// MultiHeadAttention_41678362640691
// MI455X (gfx1250) — matrix operand order repaired and checked by reading, not compiled, not run on hardware
//
#include <hip/hip_runtime.h>
#include <hip/hip_bf16.h>

#define DMODEL 1024
#define SEQ    2048
#define BATCH  2
#define HEADS  16
#define HDIM   64
#define MTOT   (BATCH * SEQ)   // 4096

typedef __attribute__((ext_vector_type(16))) __bf16 v16bf;
typedef __attribute__((ext_vector_type(8)))  float  v8f;
typedef unsigned short u16;
typedef unsigned int   u32;

static __device__ __forceinline__ u16 f2bf(float f) {
    u32 u = __float_as_uint(f);
    u32 r = u + 0x7FFFu + ((u >> 16) & 1u);   // round-to-nearest-even
    return (u16)(r >> 16);
}

union Frag16 { v16bf v; uint4 q[2]; u16 s[16]; };

// A-layout fragment (16x32 bf16): lane holds row m=lane&15; elements 0..7 at
// k = 8*lane_hi + e, elements 8..15 at k = 16 + 8*lane_hi + (e-8).
// Caller passes p already offset by 8*lane_hi; chunks at p and p+16 (ushorts).
static __device__ __forceinline__ v16bf load_a_frag(const u16* p) {
    Frag16 f;
    f.q[0] = *reinterpret_cast<const uint4*>(p);
    f.q[1] = *reinterpret_cast<const uint4*>(p + 16);
    return f.v;
}
// B-layout fragment (32x16 bf16): lane holds col n=lane&15; elements 0..7 at
// k = 8*lane_hi + e, elements 8..15 at k = 16 + 8*lane_hi + (e-8). Caller passes p already offset by 8*lane_hi.
static __device__ __forceinline__ v16bf load_b_frag(const u16* p) {
    Frag16 f;
    f.q[0] = *reinterpret_cast<const uint4*>(p);
    f.q[1] = *reinterpret_cast<const uint4*>(p + 16);
    return f.v;
}

static __device__ __forceinline__ v8f wmma_bf16(v16bf a, v16bf b, v8f c) {
    return __builtin_amdgcn_wmma_f32_16x16x32_bf16(false, a, false, b, (short)0, c, false, false);
}

// ---------------------------------------------------------------- convert x
__global__ void convert_inputs_kernel(const float* __restrict__ q,
                                      const float* __restrict__ k,
                                      const float* __restrict__ v,
                                      u16* __restrict__ xout) {
    const float* src = (blockIdx.z == 0) ? q : (blockIdx.z == 1) ? k : v;
    u16* dst = xout + (size_t)blockIdx.z * ((size_t)MTOT * DMODEL);
    int i = (blockIdx.x * blockDim.x + threadIdx.x) * 4;
    float4 f = *reinterpret_cast<const float4*>(src + i);
    ushort4 o;
    o.x = f2bf(f.x); o.y = f2bf(f.y); o.z = f2bf(f.z); o.w = f2bf(f.w);
    *reinterpret_cast<ushort4*>(dst + i) = o;
}

// ------------------------------------------------- transpose weights -> bf16
__global__ void transpose_w_kernel(const float* __restrict__ w0,
                                   const float* __restrict__ w1,
                                   const float* __restrict__ w2,
                                   const float* __restrict__ w3,
                                   u16* __restrict__ wtout) {
    __shared__ float tile[32][33];
    int z = blockIdx.z;
    const float* w = (z == 0) ? w0 : (z == 1) ? w1 : (z == 2) ? w2 : w3;
    u16* dst = wtout + (size_t)z * DMODEL * DMODEL;
    int tx = threadIdx.x, ty = threadIdx.y;
    int x = blockIdx.x * 32 + tx;
#pragma unroll
    for (int i = 0; i < 4; ++i) {
        int y = blockIdx.y * 32 + ty + i * 8;
        tile[ty + i * 8][tx] = w[(size_t)y * DMODEL + x];
    }
    __syncthreads();
#pragma unroll
    for (int i = 0; i < 4; ++i) {
        int n  = blockIdx.x * 32 + ty + i * 8;
        int kk = blockIdx.y * 32 + tx;
        dst[(size_t)n * DMODEL + kk] = f2bf(tile[tx][ty + i * 8]);
    }
}

// --------------------------------------------------------------- bf16 GEMM
// C[4096,1024] = A[4096,1024] * B + bias, B given as BT[N][K] bf16.
// Block = 128 thr = 4 waves; wave tile 32M x 64N (2 A-frags x 4 B-frags =
// 8 WMMA per 32-deep K-chunk), register double-buffered. launch_bounds(.,1)
// releases the full VGPR file (one fat wave per SIMD, no spills) so the
// double buffer lives entirely in registers.
__global__ void __launch_bounds__(128, 1)
gemm_bf16_kernel(const u16* __restrict__ A,
                 const u16* __restrict__ BT,
                 const float* __restrict__ bias,
                 u16* __restrict__ outBf,
                 float* __restrict__ outF,
                 int mode) {
    const int tid  = threadIdx.x;
    const int wave = tid >> 5;
    const int lane = tid & 31;
    const int l15  = lane & 15;
    const int hi   = (lane >> 4) & 1;

    const int m0 = blockIdx.y * 128 + wave * 32;
    const int n0 = blockIdx.x * 64;

    const v8f vzero = {0.f, 0.f, 0.f, 0.f, 0.f, 0.f, 0.f, 0.f};
    v8f acc[2][4];
#pragma unroll
    for (int mt = 0; mt < 2; ++mt)
#pragma unroll
        for (int t = 0; t < 4; ++t) acc[mt][t] = vzero;

    const u16* arow0 = A + (size_t)(m0 + l15) * DMODEL + hi * 8;
    const u16* arow1 = arow0 + (size_t)16 * DMODEL;
    const u16* brow[4];
#pragma unroll
    for (int t = 0; t < 4; ++t)
        brow[t] = BT + (size_t)(n0 + t * 16 + l15) * DMODEL + hi * 8;

    // ---- prologue: load chunk 0
    v16bf a0c = load_a_frag(arow0);
    v16bf a1c = load_a_frag(arow1);
    v16bf bc[4];
#pragma unroll
    for (int t = 0; t < 4; ++t) bc[t] = load_b_frag(brow[t]);

    for (int kb = 0; kb < DMODEL - 32; kb += 32) {
        // prefetch far-ahead A stream (gfx1250 global_prefetch_b8)
        __builtin_prefetch(arow0 + kb + 256, 0, 1);
        __builtin_prefetch(arow1 + kb + 256, 0, 1);
        // issue next chunk's loads
        v16bf a0n = load_a_frag(arow0 + kb + 32);
        v16bf a1n = load_a_frag(arow1 + kb + 32);
        v16bf bn[4];
#pragma unroll
        for (int t = 0; t < 4; ++t) bn[t] = load_b_frag(brow[t] + kb + 32);
        // compute on current chunk while next loads are in flight
#pragma unroll
        for (int t = 0; t < 4; ++t) {
            acc[0][t] = wmma_bf16(a0c, bc[t], acc[0][t]);
            acc[1][t] = wmma_bf16(a1c, bc[t], acc[1][t]);
        }
        a0c = a0n; a1c = a1n;
#pragma unroll
        for (int t = 0; t < 4; ++t) bc[t] = bn[t];
    }
    // ---- epilogue chunk
#pragma unroll
    for (int t = 0; t < 4; ++t) {
        acc[0][t] = wmma_bf16(a0c, bc[t], acc[0][t]);
        acc[1][t] = wmma_bf16(a1c, bc[t], acc[1][t]);
    }

    // ---- store
#pragma unroll
    for (int mt = 0; mt < 2; ++mt) {
#pragma unroll
        for (int t = 0; t < 4; ++t) {
            const int n = n0 + t * 16 + l15;
            const float bs = bias[n];
#pragma unroll
            for (int r = 0; r < 8; ++r) {
                const int m = m0 + mt * 16 + r + hi * 8;
                const float val = acc[mt][t][r] + bs;
                if (mode == 0) {
                    outBf[(size_t)m * DMODEL + n] = f2bf(val);
                } else if (mode == 1) {
                    const int bb = m >> 11, s = m & (SEQ - 1);
                    const int h = n >> 6, d = n & 63;
                    outBf[(((size_t)(bb * HEADS + h) * HDIM + d) << 11) + s] = f2bf(val);
                } else {
                    outF[(size_t)m * DMODEL + n] = val;
                }
            }
        }
    }
}

// ---------------------------------------------------------- flash attention
// grid (S/64, B*H), block 128 (4 waves). Wave owns 16 q rows.
// Scores computed TRANSPOSED (S^T = K_blk * Q^T): the D-fragment of S^T is
// elementwise identical to the A-fragment layout required for P*V, so no LDS
// transpose is needed. Softmax rows live per-lane (q = lane&15); halves are
// combined with one shfl_xor(16); per-row corr/norm gathered with bpermute.
__global__ void __launch_bounds__(128, 1)
attention_kernel(const u16* __restrict__ qbuf,
                 const u16* __restrict__ kbuf,
                 const u16* __restrict__ vtbuf,
                 u16* __restrict__ outb) {
    const int tid  = threadIdx.x;
    const int wave = tid >> 5;
    const int lane = tid & 31;
    const int l15  = lane & 15;
    const int hi   = (lane >> 4) & 1;

    const int bh = blockIdx.y;
    const int b  = bh >> 4;
    const int h  = bh & 15;
    const int qbase = blockIdx.x * 64 + wave * 16;

    const u16* Qh = qbuf + (size_t)b * SEQ * DMODEL + h * HDIM;
    const u16* Kh = kbuf + (size_t)b * SEQ * DMODEL + h * HDIM;
    const u16* Vt = vtbuf + (size_t)(b * HEADS + h) * HDIM * SEQ;

    // Q as B-operand of S^T = K * Q^T, fixed for the whole loop:
    const u16* qrow = Qh + (size_t)(qbase + l15) * DMODEL + hi * 8;
    const v16bf bq0 = load_b_frag(qrow);
    const v16bf bq1 = load_b_frag(qrow + 32);

    const v8f vzero = {0.f, 0.f, 0.f, 0.f, 0.f, 0.f, 0.f, 0.f};
    v8f o[4];
#pragma unroll
    for (int t = 0; t < 4; ++t) o[t] = vzero;

    float mrun = -1e30f, lrun = 0.f;   // per-lane: softmax row q = qbase+l15
    const int kend = qbase + 16;       // causal: keys 0 .. qbase+15

    for (int kb = 0; kb < kend; kb += 32) {
        const u16* krow0 = Kh + (size_t)(kb + l15) * DMODEL + hi * 8;
        const u16* krow1 = Kh + (size_t)(kb + 16 + l15) * DMODEL + hi * 8;
        v16bf ak00 = load_a_frag(krow0);
        v16bf ak01 = load_a_frag(krow0 + 32);
        v16bf ak10 = load_a_frag(krow1);
        v16bf ak11 = load_a_frag(krow1 + 32);
        v16bf bv[4];
#pragma unroll
        for (int t = 0; t < 4; ++t)
            bv[t] = load_b_frag(Vt + (size_t)(t * 16 + l15) * SEQ + kb + hi * 8);
        __builtin_prefetch(krow0 + 32 * DMODEL, 0, 1);
        __builtin_prefetch(krow1 + 32 * DMODEL, 0, 1);

        // ---- S^T tiles: rows = keys, cols = q
        v8f s0 = vzero, s1 = vzero;
        s0 = wmma_bf16(ak00, bq0, s0);
        s0 = wmma_bf16(ak01, bq1, s0);
        s1 = wmma_bf16(ak10, bq0, s1);
        s1 = wmma_bf16(ak11, bq1, s1);

        // ---- scale + causal mask. Element r: key = kb + {0,16} + r + 8*hi,
        //      q = qbase + l15.
        const int qg = qbase + l15;
#pragma unroll
        for (int r = 0; r < 8; ++r) {
            float v0 = s0[r] * 0.125f;
            float v1 = s1[r] * 0.125f;
            if (kb + r + hi * 8 > qg)      v0 = -1e30f;
            if (kb + 16 + r + hi * 8 > qg) v1 = -1e30f;
            s0[r] = v0; s1[r] = v1;
        }

        // ---- online softmax: 16 values per lane + combine halves once
        float mx = -1e30f;
#pragma unroll
        for (int r = 0; r < 8; ++r) mx = fmaxf(mx, fmaxf(s0[r], s1[r]));
        mx = fmaxf(mx, __shfl_xor(mx, 16));
        const float nm   = fmaxf(mrun, mx);
        const float corr = __expf(mrun - nm);
        float rs = 0.f;
#pragma unroll
        for (int r = 0; r < 8; ++r) {
            s0[r] = __expf(s0[r] - nm);
            s1[r] = __expf(s1[r] - nm);
            rs += s0[r] + s1[r];
        }
        rs += __shfl_xor(rs, 16);
        lrun = lrun * corr + rs;
        mrun = nm;

        // ---- P already in A-layout: e(0..7)=s0[e], e(8..15)=s1[e-8]
        Frag16 pf;
#pragma unroll
        for (int r = 0; r < 8; ++r) {
            pf.s[r]     = f2bf(s0[r]);
            pf.s[8 + r] = f2bf(s1[r]);
        }

        // ---- rescale O by per-row corr (row of o element r is q=r+8*hi)
#pragma unroll
        for (int r = 0; r < 8; ++r) {
            const float cb = __shfl(corr, r + 8 * hi);
            o[0][r] *= cb; o[1][r] *= cb; o[2][r] *= cb; o[3][r] *= cb;
        }

        // ---- O += P * V
#pragma unroll
        for (int t = 0; t < 4; ++t) o[t] = wmma_bf16(pf.v, bv[t], o[t]);
    }

    // ---- normalize (gather 1/l per row) and store row-major bf16
    const float inv = 1.0f / lrun;
#pragma unroll
    for (int r = 0; r < 8; ++r) {
        const float ib = __shfl(inv, r + 8 * hi);
        o[0][r] *= ib; o[1][r] *= ib; o[2][r] *= ib; o[3][r] *= ib;
    }
#pragma unroll
    for (int t = 0; t < 4; ++t) {
#pragma unroll
        for (int r = 0; r < 8; ++r) {
            const int qg = qbase + r + hi * 8;
            outb[((size_t)b * SEQ + qg) * DMODEL + h * HDIM + t * 16 + l15] =
                f2bf(o[t][r]);
        }
    }
}

// ---------------------------------------------------------------- launcher
extern "C" void kernel_launch(void* const* d_in, const int* in_sizes, int n_in,
                              void* d_out, int out_size, void* d_ws, size_t ws_size,
                              hipStream_t stream) {
    const float* in_q = (const float*)d_in[0];
    const float* in_k = (const float*)d_in[1];
    const float* in_v = (const float*)d_in[2];
    const float* w_q  = (const float*)d_in[4];
    const float* b_q  = (const float*)d_in[5];
    const float* w_k  = (const float*)d_in[6];
    const float* b_k  = (const float*)d_in[7];
    const float* w_v  = (const float*)d_in[8];
    const float* b_v  = (const float*)d_in[9];
    const float* w_o  = (const float*)d_in[10];
    const float* b_o  = (const float*)d_in[11];

    const size_t ACT = (size_t)MTOT * DMODEL;     // 4M elements
    const size_t WSZ = (size_t)DMODEL * DMODEL;   // 1M elements

    char* ws = (char*)d_ws;
    u16* xbuf = (u16*)ws;                         // 3 * ACT bf16
    u16* wT   = xbuf + 3 * ACT;                   // 4 * WSZ bf16
    u16* qb   = wT + 4 * WSZ;                     // ACT bf16
    u16* kb   = qb + ACT;                         // ACT bf16
    u16* vt   = kb + ACT;                         // ACT bf16 (per-head transposed)
    u16* attn = vt + ACT;                         // ACT bf16

    convert_inputs_kernel<<<dim3((unsigned)(ACT / (256 * 4)), 1, 3), 256, 0, stream>>>(
        in_q, in_k, in_v, xbuf);

    transpose_w_kernel<<<dim3(32, 32, 4), dim3(32, 8), 0, stream>>>(
        w_q, w_k, w_v, w_o, wT);

    gemm_bf16_kernel<<<dim3(16, 32), 128, 0, stream>>>(
        xbuf,           wT,           b_q, qb, nullptr, 0);
    gemm_bf16_kernel<<<dim3(16, 32), 128, 0, stream>>>(
        xbuf + ACT,     wT + WSZ,     b_k, kb, nullptr, 0);
    gemm_bf16_kernel<<<dim3(16, 32), 128, 0, stream>>>(
        xbuf + 2 * ACT, wT + 2 * WSZ, b_v, vt, nullptr, 1);

    attention_kernel<<<dim3(SEQ / 64, BATCH * HEADS), 128, 0, stream>>>(
        qb, kb, vt, attn);

    gemm_bf16_kernel<<<dim3(16, 32), 128, 0, stream>>>(
        attn, wT + 3 * WSZ, b_o, nullptr, (float*)d_out, 2);
}
